// MultiHeadAttention_34909494182535
// MI455X (gfx1250) — matrix operand order repaired and checked by reading, not compiled, not run on hardware
//
#include <hip/hip_runtime.h>
#include <hip/hip_bf16.h>

// ---------------------------------------------------------------------------
// MHA forward for MI455X (gfx1250, wave32, WMMA + TDM).
//   B=4, T=2048, C=1024, H=16, D=64.
//   0) one-shot: x -> bf16; each W -> bf16 transposed [N][K]
//   1) 3x GEMM (TDM double-buffered LDS staging, v_wmma_f32_16x16x32_bf16)
//   2) flash attention (scores = K.Q^T * C^-0.5, causal, online softmax)
//   3) GEMM (attn @ Wproj + bias) -> f32 output
// Workspace: xb 16MB + 4xWt 8MB + K/Q/V 48MB + attn 16MB = 88 MB.
// ---------------------------------------------------------------------------

typedef __attribute__((ext_vector_type(16))) __bf16        v16bf;
typedef __attribute__((ext_vector_type(8)))  __bf16        v8bf;
typedef __attribute__((ext_vector_type(8)))  float         v8f;
typedef __attribute__((ext_vector_type(4)))  unsigned int  v4u;
typedef __attribute__((ext_vector_type(8)))  int           v8i;
typedef __attribute__((ext_vector_type(4)))  int           v4i;

#define WMMA_BF16(A, Bm, Cm) \
  __builtin_amdgcn_wmma_f32_16x16x32_bf16(false, (A), false, (Bm), (short)0, (Cm), false, false)

// ---- fragment loaders (ISA 7.12.2 layouts, wave32) -------------------------
__device__ __forceinline__ v16bf frag_a16x32(const __bf16* base, int stride,
                                             int mbase, int kk, int lane) {
  int row = mbase + (lane & 15);
  int h8  = (lane >> 4) << 3;
  const __bf16* p = base + (size_t)row * stride + kk + h8;
  v8bf lo = *(const v8bf*)p;
  v8bf hi = *(const v8bf*)(p + 16);
  v16bf f;
#pragma unroll
  for (int i = 0; i < 8; ++i) { f[i] = lo[i]; f[i + 8] = hi[i]; }
  return f;
}

__device__ __forceinline__ v16bf frag_b32x16(const __bf16* Bt, int stride,
                                             int nbase, int kk, int lane) {
  int col = nbase + (lane & 15);
  int h8  = (lane >> 4) << 3;
  const __bf16* p = Bt + (size_t)col * stride + kk + h8;
  v8bf lo = *(const v8bf*)p;
  v8bf hi = *(const v8bf*)(p + 16);
  v16bf f;
#pragma unroll
  for (int i = 0; i < 8; ++i) { f[i] = lo[i]; f[i + 8] = hi[i]; }
  return f;
}

// ---- TDM: 2D bf16 tile (tile_d0 x tile_d1) -> LDS with row padding ---------
// pad_interval=4 (every 32 DWORDs = 64 bf16), pad_amount=7 (8 DWORDs = 16 bf16)
// => LDS row stride 80 bf16 when tile_d0 == 64.
__device__ __forceinline__ void tdm_load_2d(unsigned int lds_addr, const void* gaddr,
                                            unsigned int tensor_d0, unsigned int tensor_d1,
                                            unsigned int tile_d0, unsigned int tile_d1,
                                            unsigned int stride0) {
  unsigned long long ga = (unsigned long long)gaddr;
  v4u g0;
  g0[0] = 1u;                                              // count=1, user desc
  g0[1] = lds_addr;                                        // LDS byte offset
  g0[2] = (unsigned int)(ga & 0xFFFFFFFFu);                // global_addr lo
  g0[3] = (unsigned int)((ga >> 32) & 0x1FFFFFFu) | (2u << 30);  // addr hi | type=2
  v8i g1;
  g1[0] = (int)((1u << 16) |                               // data_size = 2 bytes
                (1u << 20) |                               // pad_enable
                (4u << 22) |                               // pad_interval: 32 DW
                (7u << 25));                               // pad_amount: 8 DW
  g1[1] = (int)((tensor_d0 & 0xFFFFu) << 16);              // tensor_dim0[15:0]
  g1[2] = (int)((tensor_d0 >> 16) | ((tensor_d1 & 0xFFFFu) << 16));
  g1[3] = (int)((tensor_d1 >> 16) | (tile_d0 << 16));      // tile_dim0
  g1[4] = (int)tile_d1;                                    // tile_dim1, tile_dim2=0
  g1[5] = (int)stride0;                                    // tensor_dim0_stride lo
  g1[6] = 0;
  g1[7] = 0;
  v4i z4 = {0, 0, 0, 0};
#if __clang_major__ >= 23
  v8i z8 = {0, 0, 0, 0, 0, 0, 0, 0};
  __builtin_amdgcn_tensor_load_to_lds(g0, g1, z4, z4, z8, 0);
#else
  __builtin_amdgcn_tensor_load_to_lds(g0, g1, z4, z4, 0);
#endif
}

__device__ __forceinline__ unsigned int lds_off(const void* p) {
  return (unsigned int)(size_t)p;   // generic ptr low 32 bits = LDS offset
}

// ---------------------------------------------------------------------------
// One-shot converters.
// ---------------------------------------------------------------------------
__global__ __launch_bounds__(256)
void convert_f32_bf16(const float* __restrict__ src, __bf16* __restrict__ dst) {
  size_t i = ((size_t)blockIdx.x * 256 + threadIdx.x) * 4;
  float4 v = *(const float4*)(src + i);
  __bf16* d = dst + i;
  d[0] = (__bf16)v.x; d[1] = (__bf16)v.y; d[2] = (__bf16)v.z; d[3] = (__bf16)v.w;
}

// W f32 [K][N] -> Wt bf16 [N][K], 64x64 tiles.
__global__ __launch_bounds__(256)
void transpose_convert_bf16(const float* __restrict__ W, __bf16* __restrict__ Wt,
                            int K, int N) {
  __shared__ __bf16 t[64][72];
  const int k0 = blockIdx.x * 64, n0 = blockIdx.y * 64;
  const int tid = threadIdx.x;
  for (int i = tid; i < 1024; i += 256) {
    int r = i >> 4;           // k within tile
    int c = (i & 15) << 2;    // n within tile
    float4 v = *(const float4*)(W + (size_t)(k0 + r) * N + n0 + c);
    t[c + 0][r] = (__bf16)v.x; t[c + 1][r] = (__bf16)v.y;
    t[c + 2][r] = (__bf16)v.z; t[c + 3][r] = (__bf16)v.w;
  }
  __syncthreads();
  for (int i = tid; i < 512; i += 256) {
    int r = i >> 3;           // n within tile
    int c = (i & 7) << 3;     // k within tile
    v8bf o;
#pragma unroll
    for (int e = 0; e < 8; ++e) o[e] = t[r][c + e];
    *(v8bf*)(Wt + (size_t)(n0 + r) * K + k0 + c) = o;
  }
}

// ---------------------------------------------------------------------------
// GEMM: D[M,N] = A[M,K](bf16) @ Bt[N,K](bf16)^T (+bias).
// Block tile 128x128, 8 waves, wave tile 32x64, K-stage 64.
// LDS staged by TDM, double-buffered; compute overlaps the async DMA.
// ---------------------------------------------------------------------------
#define GT_M 128
#define GT_N 128
#define GT_K 64
#define LDS_AS 80   // = 64 + TDM pad (8 DWORDs); rows stay 16B aligned

template <bool OUT_F32>
__global__ __launch_bounds__(256)
void gemm_tdm_wmma(const __bf16* __restrict__ A, const __bf16* __restrict__ Bt,
                   const float* __restrict__ bias, void* __restrict__ Dptr,
                   int M, int N, int K) {
  __shared__ __bf16 As[2][GT_M * LDS_AS];
  __shared__ __bf16 Bs[2][GT_N * LDS_AS];

  const int tid  = threadIdx.x;
  const int lane = tid & 31;
  const int wid  = tid >> 5;
  const int waveM = wid >> 1;
  const int waveN = wid & 1;
  const int m0 = blockIdx.x * GT_M;
  const int n0 = blockIdx.y * GT_N;

  v8f acc[2][4] = {};

  if (wid == 0) {
    tdm_load_2d(lds_off(&As[0][0]), A + (size_t)m0 * K, K, M, GT_K, GT_M, K);
    tdm_load_2d(lds_off(&Bs[0][0]), Bt + (size_t)n0 * K, K, N, GT_K, GT_N, K);
  }
  __builtin_amdgcn_s_wait_tensorcnt(0);
  __syncthreads();

  int cur = 0;
  for (int k0 = 0; k0 < K; k0 += GT_K) {
    int nxt = cur ^ 1;
    if ((k0 + GT_K < K) && wid == 0) {
      tdm_load_2d(lds_off(&As[nxt][0]), A + (size_t)m0 * K + (k0 + GT_K), K, M, GT_K, GT_M, K);
      tdm_load_2d(lds_off(&Bs[nxt][0]), Bt + (size_t)n0 * K + (k0 + GT_K), K, N, GT_K, GT_N, K);
    }
#pragma unroll
    for (int kk = 0; kk < GT_K; kk += 32) {
      v16bf af[2], bfm[4];
#pragma unroll
      for (int mt = 0; mt < 2; ++mt)
        af[mt] = frag_a16x32(&As[cur][0], LDS_AS, waveM * 32 + mt * 16, kk, lane);
#pragma unroll
      for (int nt = 0; nt < 4; ++nt)
        bfm[nt] = frag_b32x16(&Bs[cur][0], LDS_AS, waveN * 64 + nt * 16, kk, lane);
#pragma unroll
      for (int mt = 0; mt < 2; ++mt)
#pragma unroll
        for (int nt = 0; nt < 4; ++nt)
          acc[mt][nt] = WMMA_BF16(af[mt], bfm[nt], acc[mt][nt]);
    }
    __builtin_amdgcn_s_wait_tensorcnt(0);   // next-stage DMA done (issuing wave)
    __syncthreads();                        // all waves done reading `cur`
    cur = nxt;
  }

  const int lm = (lane >> 4) << 3;
  const int ln = lane & 15;
#pragma unroll
  for (int mt = 0; mt < 2; ++mt) {
#pragma unroll
    for (int nt = 0; nt < 4; ++nt) {
#pragma unroll
      for (int r = 0; r < 8; ++r) {
        int row = m0 + waveM * 32 + mt * 16 + lm + r;
        int col = n0 + waveN * 64 + nt * 16 + ln;
        if (OUT_F32) {
          ((float*)Dptr)[(size_t)row * N + col] = acc[mt][nt][r] + bias[col];
        } else {
          ((__bf16*)Dptr)[(size_t)row * N + col] = (__bf16)acc[mt][nt][r];
        }
      }
    }
  }
}

// ---------------------------------------------------------------------------
// Flash attention (causal). Reference quirk: wei = K @ Q^T * C^-0.5.
// One block per (b, h, 64-row strip); 4 waves x 16 rows.
// ---------------------------------------------------------------------------
#define FA_T 2048
#define FA_C 1024
#define FA_H 16
#define FA_D 64
#define QS_S 80
#define VT_S 40

__global__ __launch_bounds__(128)
void flash_attn_wmma(const __bf16* __restrict__ kb, const __bf16* __restrict__ qb,
                     const __bf16* __restrict__ vb, __bf16* __restrict__ ob) {
  const int nIB = FA_T / 64;
  const int ib = blockIdx.x % nIB;
  const int h  = (blockIdx.x / nIB) % FA_H;
  const int b  = blockIdx.x / (nIB * FA_H);
  const int i0 = ib * 64;

  const int tid  = threadIdx.x;
  const int wave = tid >> 5;
  const int lane = tid & 31;
  const int i_base = i0 + wave * 16;
  const int lm = (lane >> 4) << 3;
  const int ln = lane & 15;

  const size_t bh = (size_t)b * FA_T * FA_C + h * FA_D;
  const __bf16* Kg = kb + bh;
  const __bf16* Qg = qb + bh;
  const __bf16* Vg = vb + bh;

  __shared__ __bf16 Qs[32 * QS_S];
  __shared__ __bf16 Vt[FA_D * VT_S];
  __shared__ __bf16 Ps[4][16 * VT_S];

  v16bf a0 = frag_a16x32(Kg, FA_C, i_base, 0, lane);
  v16bf a1 = frag_a16x32(Kg, FA_C, i_base, 32, lane);

  v8f oacc[4] = {};
  float mr[8], lr[8];
#pragma unroll
  for (int r = 0; r < 8; ++r) { mr[r] = -1e30f; lr[r] = 0.f; }

  const float scale = 0.03125f;
  const int j_end = i0 + 64;

  for (int j0 = 0; j0 < j_end; j0 += 32) {
    for (int i = tid; i < 256; i += 128) {
      int row = i >> 3, col = (i & 7) << 3;
      *(v8bf*)&Qs[row * QS_S + col] =
          *(const v8bf*)(Qg + (size_t)(j0 + row) * FA_C + col);
    }
    for (int i = tid; i < 256; i += 128) {
      int row = i >> 3, col = (i & 7) << 3;
      v8bf v = *(const v8bf*)(Vg + (size_t)(j0 + row) * FA_C + col);
#pragma unroll
      for (int e = 0; e < 8; ++e) Vt[(col + e) * VT_S + row] = v[e];
    }
    if (j0 + 32 < j_end) {   // hint next tiles toward L2 (global_prefetch_b8)
      __builtin_prefetch(Qg + (size_t)(j0 + 32 + (tid & 31)) * FA_C, 0, 1);
      __builtin_prefetch(Vg + (size_t)(j0 + 32 + (tid & 31)) * FA_C, 0, 1);
    }
    __syncthreads();

    if (j0 <= i_base + 15) {
      v8f s[2] = {};
#pragma unroll
      for (int jt = 0; jt < 2; ++jt) {
        v16bf b0 = frag_b32x16(Qs, QS_S, jt * 16, 0, lane);
        v16bf b1 = frag_b32x16(Qs, QS_S, jt * 16, 32, lane);
        s[jt] = WMMA_BF16(a0, b0, s[jt]);
        s[jt] = WMMA_BF16(a1, b1, s[jt]);
      }
#pragma unroll
      for (int r = 0; r < 8; ++r) {
        int irow = i_base + lm + r;
        float v0 = s[0][r] * scale;
        float v1 = s[1][r] * scale;
        if (j0 + ln > irow)      v0 = -1e30f;
        if (j0 + 16 + ln > irow) v1 = -1e30f;
        float mx = fmaxf(v0, v1);
#pragma unroll
        for (int off = 1; off < 16; off <<= 1)
          mx = fmaxf(mx, __shfl_xor(mx, off, 16));
        float nm = fmaxf(mr[r], mx);
        float al = __expf(mr[r] - nm);
        float p0 = __expf(v0 - nm);
        float p1 = __expf(v1 - nm);
        float rs = p0 + p1;
#pragma unroll
        for (int off = 1; off < 16; off <<= 1)
          rs += __shfl_xor(rs, off, 16);
        lr[r] = lr[r] * al + rs;
        mr[r] = nm;
        oacc[0][r] *= al; oacc[1][r] *= al;
        oacc[2][r] *= al; oacc[3][r] *= al;
        Ps[wave][(lm + r) * VT_S + ln]      = (__bf16)p0;
        Ps[wave][(lm + r) * VT_S + 16 + ln] = (__bf16)p1;
      }
      v16bf pf = frag_a16x32(Ps[wave], VT_S, 0, 0, lane);
#pragma unroll
      for (int nt = 0; nt < 4; ++nt) {
        v16bf vf = frag_b32x16(Vt, VT_S, nt * 16, 0, lane);
        oacc[nt] = WMMA_BF16(pf, vf, oacc[nt]);
      }
    }
    __syncthreads();
  }

  __bf16* Og = ob + bh;
#pragma unroll
  for (int r = 0; r < 8; ++r) {
    float inv = 1.0f / lr[r];
    int irow = i_base + lm + r;
#pragma unroll
    for (int nt = 0; nt < 4; ++nt)
      Og[(size_t)irow * FA_C + nt * 16 + ln] = (__bf16)(oacc[nt][r] * inv);
  }
}

// ---------------------------------------------------------------------------
extern "C" void kernel_launch(void* const* d_in, const int* in_sizes, int n_in,
                              void* d_out, int out_size, void* d_ws, size_t ws_size,
                              hipStream_t stream) {
  const float* x     = (const float*)d_in[0];
  const float* Wk    = (const float*)d_in[1];
  const float* Wq    = (const float*)d_in[2];
  const float* Wv    = (const float*)d_in[3];
  const float* Wproj = (const float*)d_in[4];
  const float* bproj = (const float*)d_in[5];

  const int M = 4 * 2048, N = 1024, K = 1024;
  const size_t mat = (size_t)M * N;    // 8.4M
  const size_t wmat = (size_t)K * N;   // 1M

  __bf16* xb  = (__bf16*)d_ws;         // 16 MB
  __bf16* Wkt = xb + mat;              // 2 MB each, transposed [N][K]
  __bf16* Wqt = Wkt + wmat;
  __bf16* Wvt = Wqt + wmat;
  __bf16* Wpt = Wvt + wmat;
  __bf16* kb  = Wpt + wmat;            // 16 MB each
  __bf16* qb  = kb + mat;
  __bf16* vb  = qb + mat;
  __bf16* ab  = vb + mat;

  // one-shot converts
  convert_f32_bf16<<<(unsigned)(mat / 4 / 256), 256, 0, stream>>>(x, xb);
  dim3 tg(K / 64, N / 64);
  transpose_convert_bf16<<<tg, 256, 0, stream>>>(Wk, Wkt, K, N);
  transpose_convert_bf16<<<tg, 256, 0, stream>>>(Wq, Wqt, K, N);
  transpose_convert_bf16<<<tg, 256, 0, stream>>>(Wv, Wvt, K, N);
  transpose_convert_bf16<<<tg, 256, 0, stream>>>(Wproj, Wpt, K, N);

  dim3 gg(M / GT_M, N / GT_N);
  gemm_tdm_wmma<false><<<gg, 256, 0, stream>>>(xb, Wkt, nullptr, kb, M, N, K);
  gemm_tdm_wmma<false><<<gg, 256, 0, stream>>>(xb, Wqt, nullptr, qb, M, N, K);
  gemm_tdm_wmma<false><<<gg, 256, 0, stream>>>(xb, Wvt, nullptr, vb, M, N, K);

  flash_attn_wmma<<<4 * FA_H * (FA_T / 64), 128, 0, stream>>>(kb, qb, vb, ab);

  gemm_tdm_wmma<true><<<gg, 256, 0, stream>>>(ab, Wpt, bproj, d_out, M, N, K);
}
